// VoxelBackBone8x_16707422782122
// MI455X (gfx1250) — compile-verified, matrix operand order checked by reading, not run on hardware
//
#include <hip/hip_runtime.h>

typedef __attribute__((ext_vector_type(16))) _Float16 v16h;
typedef __attribute__((ext_vector_type(8)))  _Float16 v8h;
typedef __attribute__((ext_vector_type(8)))  float    v8f;

__device__ __forceinline__ v8h zero_v8h() {
  v8h z = {(_Float16)0.f, (_Float16)0.f, (_Float16)0.f, (_Float16)0.f,
           (_Float16)0.f, (_Float16)0.f, (_Float16)0.f, (_Float16)0.f};
  return z;
}

// ---------------- elementwise helper kernels ----------------

__global__ void zero_f16_kernel(_Float16* __restrict__ p, int n) {
  int i = blockIdx.x * blockDim.x + threadIdx.x;
  if (i < n) p[i] = (_Float16)0.f;
}

__global__ void zero_u8_kernel(unsigned char* __restrict__ p, int n) {
  int i = blockIdx.x * blockDim.x + threadIdx.x;
  if (i < n) p[i] = 0;
}

__global__ void scatter_kernel(const float* __restrict__ feat, const int* __restrict__ coords,
                               int npts, _Float16* __restrict__ buf,
                               unsigned char* __restrict__ mask,
                               int zd, int yd, int xd, int cin) {
  int i = blockIdx.x * blockDim.x + threadIdx.x;
  if (i >= npts) return;
  int b = coords[i * 4 + 0], z = coords[i * 4 + 1];
  int y = coords[i * 4 + 2], x = coords[i * 4 + 3];
  int v = ((b * zd + z) * yd + y) * xd + x;
  mask[v] = 1;
  for (int c = 0; c < cin; ++c) buf[v * cin + c] = (_Float16)feat[i * cin + c];
}

// fp32 (kz,ky,kx,cin,cout) -> fp16 [cout][Kpad] (K = ((kz*KY+ky)*KX+kx)*cin+ci), zero-padded
__global__ void wprep_kernel(const float* __restrict__ w, _Float16* __restrict__ wt,
                             int Ktot, int Kpad, int cout) {
  int i = blockIdx.x * blockDim.x + threadIdx.x;
  if (i >= cout * Kpad) return;
  int co = i / Kpad, k = i - co * Kpad;
  wt[i] = (k < Ktot) ? (_Float16)w[k * cout + co] : (_Float16)0.f;
}

// decode table: k -> (precomputed element offset, dz, dy, dx)
// off = (dz*iyd*ixd + dy*ixd + dx)*cin + ci, so inner-loop address math is one add.
__global__ void ktab_kernel(int4* __restrict__ tab, int Ktot, int cin, int kyd, int kxd,
                            int iyd, int ixd) {
  int k = blockIdx.x * blockDim.x + threadIdx.x;
  if (k >= Ktot) return;
  int ci = k % cin;
  int r = k / cin;
  int kx = r % kxd; r /= kxd;
  int ky = r % kyd;
  int kz = r / kyd;
  int off = (kz * iyd * ixd + ky * ixd + kx) * cin + ci;
  tab[k] = make_int4(off, kz, ky, kx);
}

// mask propagation: OR over the conv window
__global__ void maskconv_kernel(const unsigned char* __restrict__ mi, unsigned char* __restrict__ mo,
                                int izd, int iyd, int ixd, int ozd, int oyd, int oxd,
                                int kzd, int kyd, int kxd, int sz, int sy, int sx,
                                int pz, int py, int px, int Mtot) {
  int m = blockIdx.x * blockDim.x + threadIdx.x;
  if (m >= Mtot) return;
  int x = m % oxd; int r = m / oxd;
  int y = r % oyd; r /= oyd;
  int z = r % ozd; int b = r / ozd;
  int zb = z * sz - pz, yb = y * sy - py, xb = x * sx - px;
  unsigned char v = 0;
  for (int kz = 0; kz < kzd; ++kz) {
    int iz = zb + kz; if ((unsigned)iz >= (unsigned)izd) continue;
    for (int ky = 0; ky < kyd; ++ky) {
      int iy = yb + ky; if ((unsigned)iy >= (unsigned)iyd) continue;
      for (int kx = 0; kx < kxd; ++kx) {
        int ix = xb + kx; if ((unsigned)ix >= (unsigned)ixd) continue;
        v |= mi[((b * izd + iz) * iyd + iy) * ixd + ix];
      }
    }
  }
  mo[m] = v ? 1 : 0;
}

__global__ void pcoords_kernel(const int* __restrict__ coords, int n, float* __restrict__ out) {
  int i = blockIdx.x * blockDim.x + threadIdx.x;
  if (i >= n) return;
  int b = coords[i * 4 + 0], z = coords[i * 4 + 1];
  int y = coords[i * 4 + 2], x = coords[i * 4 + 3];
  out[i * 4 + 0] = (float)b;
  out[i * 4 + 1] = ((float)x + 0.5f) * 0.05f - 3.2f;
  out[i * 4 + 2] = ((float)y + 0.5f) * 0.05f - 3.2f;
  out[i * 4 + 3] = ((float)z + 0.5f) * 0.1f  - 3.0f;
}

__global__ void pack_kernel(const _Float16* __restrict__ buf, const unsigned char* __restrict__ mask,
                            float* __restrict__ outF, float* __restrict__ outM, int nvox, int C) {
  int i = blockIdx.x * blockDim.x + threadIdx.x;
  int tot = nvox * C;
  if (i < tot) outF[i] = (float)buf[i];
  if (i < nvox) outM[i] = mask[i] ? 1.f : 0.f;
}

// ---------------- WMMA implicit-GEMM conv3d + BN + LeakyReLU + mask ----------------
// One wave computes a 16(out-voxel) x 16*NT(cout) tile, all fragments built
// directly in registers (no LDS, no barriers).
//
// A fragment (CDNA5 16-bit A layout): lane<16 holds row r, K{0..7,16..23};
// lane>=16 holds row r, K{8..15,24..31}. With K ordered (tap, cin) and cin a
// multiple of 8, each 8-K half is one kernel tap + contiguous channels -> one
// bounds check + one 16B global_load_b128 per half ("fastA"). cin=4 (layer 0)
// falls back to per-element gather.
//
// oxd==oyd and cout are powers of two -> decode via shifts (oxSh, coutSh);
// only the Z decode needs one integer divide per lane at setup.

template <int NT>
__launch_bounds__(128)
__global__ void conv_wmma_kernel(
    const _Float16* __restrict__ in, const _Float16* __restrict__ wt,
    const int4* __restrict__ ktab, const unsigned char* __restrict__ omask,
    const float* __restrict__ bng, const float* __restrict__ bnb,
    const float* __restrict__ bnm, const float* __restrict__ bnv,
    _Float16* __restrict__ out,
    int cin, int izd, int iyd, int ixd,
    int ozd, int oxSh, int coutSh,
    int sz, int sy, int sx, int pz, int py, int px,
    int Ktot, int Kpad, int Mtot, int fastA)
{
  const int lane  = threadIdx.x;
  const int wid   = threadIdx.y;
  const int mtile = blockIdx.x * 4 + wid;
  const int r = lane & 15;   // row (A) / column (B) index
  const int h = lane >> 4;   // half-wave selector for K interleave
  const int oxMask = (1 << oxSh) - 1;

  // decode my output voxel (row r of this M tile); x/y via shifts, z via one div
  int m = mtile * 16 + r;
  bool mvalid = (m < Mtot);
  int mc = mvalid ? m : 0;
  int xo = mc & oxMask; int t0 = mc >> oxSh;
  int yo = t0 & oxMask; t0 >>= oxSh;
  int bb = t0 / ozd;
  int zo = t0 - bb * ozd;
  int zb = zo * sz - pz, yb = yo * sy - py, xb = xo * sx - px;
  // per-lane voxel base element offset (may be "negative-ish"; only used when
  // the per-tap bounds check passes, which makes base+off valid)
  int laneBase = (((bb * izd + zb) * iyd + yb) * ixd + xb) * cin;

  // my output channels (column r of each N tile) + BN constants
  int co[NT];
  float gs[NT], bs[NT], mus[NT];
#pragma unroll
  for (int t = 0; t < NT; ++t) {
    co[t]  = (blockIdx.y * NT + t) * 16 + r;
    gs[t]  = bng[co[t]] * rsqrtf(bnv[co[t]] + 0.001f);
    bs[t]  = bnb[co[t]];
    mus[t] = bnm[co[t]];
  }

  v8f acc[NT];
#pragma unroll
  for (int t = 0; t < NT; ++t)
    acc[t] = (v8f){0.f, 0.f, 0.f, 0.f, 0.f, 0.f, 0.f, 0.f};

  const int nK = Kpad >> 5;
  for (int kc = 0; kc < nK; ++kc) {
    // ---- A fragment, straight into registers ----
    v16h afrag;
    if (fastA) {
      v8h p0 = zero_v8h(), p1 = zero_v8h();
#pragma unroll
      for (int part = 0; part < 2; ++part) {
        int ko = kc * 32 + part * 16 + h * 8;   // 8 consecutive K, one tap
        v8h v = zero_v8h();
        if (mvalid && ko < Ktot) {
          int4 kd = ktab[ko];
          int izv = zb + kd.y, iyv = yb + kd.z, ixv = xb + kd.w;
          if ((unsigned)izv < (unsigned)izd && (unsigned)iyv < (unsigned)iyd &&
              (unsigned)ixv < (unsigned)ixd) {
            v = *(const v8h*)(in + laneBase + kd.x);
          }
        }
        if (part == 0) p0 = v; else p1 = v;
      }
      afrag = __builtin_shufflevector(p0, p1, 0,1,2,3,4,5,6,7,8,9,10,11,12,13,14,15);
    } else {
#pragma unroll
      for (int part = 0; part < 2; ++part) {
#pragma unroll
        for (int t = 0; t < 8; ++t) {
          int ko = kc * 32 + part * 16 + h * 8 + t;
          _Float16 v = (_Float16)0.f;
          if (mvalid && ko < Ktot) {
            int4 kd = ktab[ko];
            int izv = zb + kd.y, iyv = yb + kd.z, ixv = xb + kd.w;
            if ((unsigned)izv < (unsigned)izd && (unsigned)iyv < (unsigned)iyd &&
                (unsigned)ixv < (unsigned)ixd) {
              v = in[laneBase + kd.x];
            }
          }
          afrag[part * 8 + t] = v;
        }
      }
    }

    if (kc + 1 < nK) __builtin_prefetch(wt + co[0] * Kpad + (kc + 1) * 32, 0, 0);

    // ---- NT WMMAs sharing one A fragment ----
#pragma unroll
    for (int t = 0; t < NT; ++t) {
      const _Float16* wk = wt + co[t] * Kpad + kc * 32;
      v8h b0 = *(const v8h*)(wk + h * 8);
      v8h b1 = *(const v8h*)(wk + 16 + h * 8);
      v16h bfrag = __builtin_shufflevector(b0, b1, 0,1,2,3,4,5,6,7,8,9,10,11,12,13,14,15);
      acc[t] = __builtin_amdgcn_wmma_f32_16x16x32_f16(false, afrag, false, bfrag,
                                                      (short)0, acc[t], false, false);
    }
  }

  // ---- epilogue: C/D layout -> lane = N, VGPR j = M row (j or j+8) ----
  int mrowBase = mtile * 16 + h * 8;
  bool mk[8];
#pragma unroll
  for (int j = 0; j < 8; ++j)
    mk[j] = (mrowBase + j < Mtot) ? (omask[mrowBase + j] != 0) : false;

#pragma unroll
  for (int t = 0; t < NT; ++t) {
#pragma unroll
    for (int j = 0; j < 8; ++j) {
      int mrow = mrowBase + j;
      if (mrow < Mtot) {
        float val = (acc[t][j] - mus[t]) * gs[t] + bs[t];
        val = (val >= 0.f) ? val : 0.01f * val;
        if (!mk[j]) val = 0.f;
        out[(mrow << coutSh) + co[t]] = (_Float16)val;
      }
    }
  }
}

// ---------------- host orchestration ----------------

static inline int ilog2(int v) { int s = 0; while ((1 << s) < v) ++s; return s; }

extern "C" void kernel_launch(void* const* d_in, const int* in_sizes, int n_in,
                              void* d_out, int out_size, void* d_ws, size_t ws_size,
                              hipStream_t stream) {
  (void)n_in; (void)out_size; (void)ws_size;
  const float* t_feat   = (const float*)d_in[0];
  const int*   t_coords = (const int*)d_in[1];
  const float* s_feat   = (const float*)d_in[2];
  const int*   s_coords = (const int*)d_in[3];
  const int n_t = in_sizes[0] / 4;
  const int n_s = in_sizes[2] / 4;
  const int NBATCH = 2;

  static const int CIN[12]  = {4,16,16,32,32,32,64,64,64,64,64,64};
  static const int COUT[12] = {16,16,32,32,32,64,64,64,64,64,64,128};
  static const int KZ[12] = {3,3,3,3,3,3,3,3,3,3,3,3};
  static const int KYd[12]= {3,3,3,3,3,3,3,3,3,3,3,1};
  static const int KXd[12]= {3,3,3,3,3,3,3,3,3,3,3,1};
  static const int SZ[12] = {1,1,2,1,1,2,1,1,2,1,1,2};
  static const int SY[12] = {1,1,2,1,1,2,1,1,2,1,1,1};
  static const int SX[12] = {1,1,2,1,1,2,1,1,2,1,1,1};
  static const int PZ[12] = {1,1,1,1,1,1,1,1,0,1,1,0};
  static const int PY[12] = {1,1,1,1,1,1,1,1,1,1,1,0};
  static const int PX[12] = {1,1,1,1,1,1,1,1,1,1,1,0};
  static const int SUBM[12]= {1,1,0,1,1,0,1,1,0,1,1,0};

  // derived per-layer geometry
  int IZ[12], IY[12], IX[12], OZ[12], OY[12], OX[12];
  int Ktot[12], Kpad[12], MtotOut[12];
  int iz = 41, iy = 128, ix = 128;
  for (int l = 0; l < 12; ++l) {
    IZ[l] = iz; IY[l] = iy; IX[l] = ix;
    OZ[l] = (iz + 2 * PZ[l] - KZ[l]) / SZ[l] + 1;
    OY[l] = (iy + 2 * PY[l] - KYd[l]) / SY[l] + 1;
    OX[l] = (ix + 2 * PX[l] - KXd[l]) / SX[l] + 1;
    Ktot[l] = KZ[l] * KYd[l] * KXd[l] * CIN[l];
    Kpad[l] = (Ktot[l] + 31) & ~31;
    MtotOut[l] = NBATCH * OZ[l] * OY[l] * OX[l];
    iz = OZ[l]; iy = OY[l]; ix = OX[l];
  }

  // workspace layout
  char* ws = (char*)d_ws;
  size_t off = 0;
  auto take = [&](size_t bytes) -> size_t {
    size_t o = off; off += (bytes + 255) & ~(size_t)255; return o;
  };
  const size_t MAXELEM = (size_t)NBATCH * 41 * 128 * 128 * 16;
  const size_t MAXVOX  = (size_t)NBATCH * 41 * 128 * 128;
  size_t offF0 = take(MAXELEM * 2);
  size_t offF1 = take(MAXELEM * 2);
  size_t offM0 = take(MAXVOX);
  size_t offM1 = take(MAXVOX);
  size_t wtOff[12], ktOff[12];
  for (int l = 0; l < 12; ++l) {
    wtOff[l] = take((size_t)Kpad[l] * COUT[l] * 2);
    ktOff[l] = take((size_t)Ktot[l] * sizeof(int4));
  }

  // per-launch weight transpose + k-decode tables
  for (int l = 0; l < 12; ++l) {
    int n = COUT[l] * Kpad[l];
    wprep_kernel<<<(n + 255) / 256, 256, 0, stream>>>(
        (const float*)d_in[4 + l], (_Float16*)(ws + wtOff[l]), Ktot[l], Kpad[l], COUT[l]);
    ktab_kernel<<<(Ktot[l] + 255) / 256, 256, 0, stream>>>(
        (int4*)(ws + ktOff[l]), Ktot[l], CIN[l], KYd[l], KXd[l], IY[l], IX[l]);
  }

  const int finV = MtotOut[11];             // 2*2*16*16 = 1024
  const size_t featN = (size_t)finV * 128;  // 131072
  float* outp = (float*)d_out;
  const float* feats[2] = {s_feat, t_feat};
  const int*   crds[2]  = {s_coords, t_coords};
  const int    npts[2]  = {n_s, n_t};
  const size_t outFOff[2] = {0, featN + (size_t)finV};
  const size_t outMOff[2] = {featN, featN + (size_t)finV + featN};

  for (int s = 0; s < 2; ++s) {
    _Float16* F0p = (_Float16*)(ws + offF0);
    _Float16* F1p = (_Float16*)(ws + offF1);
    unsigned char* M0p = (unsigned char*)(ws + offM0);
    unsigned char* M1p = (unsigned char*)(ws + offM1);
    int nvox0  = NBATCH * 41 * 128 * 128;
    int nfeat0 = nvox0 * 4;
    zero_f16_kernel<<<(nfeat0 + 255) / 256, 256, 0, stream>>>(F0p, nfeat0);
    zero_u8_kernel<<<(nvox0 + 255) / 256, 256, 0, stream>>>(M0p, nvox0);
    scatter_kernel<<<(npts[s] + 255) / 256, 256, 0, stream>>>(
        feats[s], crds[s], npts[s], F0p, M0p, 41, 128, 128, 4);

    _Float16* cur = F0p; _Float16* nxt = F1p;
    unsigned char* curM = M0p; unsigned char* othM = M1p;
    for (int l = 0; l < 12; ++l) {
      if (!SUBM[l]) {
        maskconv_kernel<<<(MtotOut[l] + 255) / 256, 256, 0, stream>>>(
            curM, othM, IZ[l], IY[l], IX[l], OZ[l], OY[l], OX[l],
            KZ[l], KYd[l], KXd[l], SZ[l], SY[l], SX[l],
            PZ[l], PY[l], PX[l], MtotOut[l]);
        unsigned char* tm = curM; curM = othM; othM = tm;
      }
      int NT = (COUT[l] >= 64) ? 4 : (COUT[l] / 16);
      int Mtiles = (MtotOut[l] + 15) / 16;
      dim3 grid((Mtiles + 3) / 4, COUT[l] / (16 * NT));
      dim3 blk(32, 4);
      int fastA = (CIN[l] % 8 == 0) ? 1 : 0;
      int oxSh = ilog2(OX[l]);
      int coutSh = ilog2(COUT[l]);
      if (NT == 1) {
        conv_wmma_kernel<1><<<grid, blk, 0, stream>>>(
            cur, (const _Float16*)(ws + wtOff[l]), (const int4*)(ws + ktOff[l]), curM,
            (const float*)d_in[16 + l], (const float*)d_in[28 + l],
            (const float*)d_in[40 + l], (const float*)d_in[52 + l],
            nxt, CIN[l], IZ[l], IY[l], IX[l], OZ[l], oxSh, coutSh,
            SZ[l], SY[l], SX[l], PZ[l], PY[l], PX[l], Ktot[l], Kpad[l], MtotOut[l], fastA);
      } else if (NT == 2) {
        conv_wmma_kernel<2><<<grid, blk, 0, stream>>>(
            cur, (const _Float16*)(ws + wtOff[l]), (const int4*)(ws + ktOff[l]), curM,
            (const float*)d_in[16 + l], (const float*)d_in[28 + l],
            (const float*)d_in[40 + l], (const float*)d_in[52 + l],
            nxt, CIN[l], IZ[l], IY[l], IX[l], OZ[l], oxSh, coutSh,
            SZ[l], SY[l], SX[l], PZ[l], PY[l], PX[l], Ktot[l], Kpad[l], MtotOut[l], fastA);
      } else {
        conv_wmma_kernel<4><<<grid, blk, 0, stream>>>(
            cur, (const _Float16*)(ws + wtOff[l]), (const int4*)(ws + ktOff[l]), curM,
            (const float*)d_in[16 + l], (const float*)d_in[28 + l],
            (const float*)d_in[40 + l], (const float*)d_in[52 + l],
            nxt, CIN[l], IZ[l], IY[l], IX[l], OZ[l], oxSh, coutSh,
            SZ[l], SY[l], SX[l], PZ[l], PY[l], PX[l], Ktot[l], Kpad[l], MtotOut[l], fastA);
      }
      _Float16* tf = cur; cur = nxt; nxt = tf;
    }
    pack_kernel<<<((int)featN + 255) / 256, 256, 0, stream>>>(
        cur, curM, outp + outFOff[s], outp + outMOff[s], finV, 128);
  }

  const size_t pcOffS = 2 * featN + 2 * (size_t)finV;
  pcoords_kernel<<<(n_s + 255) / 256, 256, 0, stream>>>(s_coords, n_s, outp + pcOffS);
  pcoords_kernel<<<(n_t + 255) / 256, 256, 0, stream>>>(t_coords, n_t,
                                                        outp + pcOffS + (size_t)n_s * 4);
}
